// ALIGNNConv_66812511256781
// MI455X (gfx1250) — matrix operand order repaired and checked by reading, not compiled, not run on hardware
//
#include <hip/hip_runtime.h>
#include <hip/hip_bf16.h>
#include <cstdint>
#include <cstddef>

typedef __bf16 bf16_t;
typedef __attribute__((ext_vector_type(16))) __bf16 v16bf;
typedef __attribute__((ext_vector_type(8)))  __bf16 v8bf;
typedef __attribute__((ext_vector_type(4)))  __bf16 v4bf;
typedef __attribute__((ext_vector_type(8)))  float  v8f;

#define FDIM 128
#define BN_EPS 1e-5f
#define GATE_EPS 1e-6f
// LDS row stride for staged W: 136 bf16 = 272 B = 68 dwords (bank-conflict-free)
#define WSTRIDE 136

static __device__ __forceinline__ v16bf mkv16(v8bf lo, v8bf hi) {
  union { v16bf v; v8bf h[2]; } u;
  u.h[0] = lo; u.h[1] = hi;
  return u.v;
}

// ---------------------------------------------------------------------------
// Elementwise helpers (vectorized, grid-stride)
// ---------------------------------------------------------------------------
__global__ __launch_bounds__(256) void f32_to_bf16_kernel(
    const float* __restrict__ in, bf16_t* __restrict__ out, size_t n4) {
  size_t i = (size_t)blockIdx.x * blockDim.x + threadIdx.x;
  size_t stride = (size_t)gridDim.x * blockDim.x;
  for (; i < n4; i += stride) {
    float4 v = ((const float4*)in)[i];
    v4bf o;
    o.x = (bf16_t)v.x; o.y = (bf16_t)v.y; o.z = (bf16_t)v.z; o.w = (bf16_t)v.w;
    ((v4bf*)out)[i] = o;
  }
}

__global__ __launch_bounds__(256) void zero_f32_kernel(
    float* __restrict__ p, size_t n4) {
  size_t i = (size_t)blockIdx.x * blockDim.x + threadIdx.x;
  size_t stride = (size_t)gridDim.x * blockDim.x;
  float4 zz = {0.f, 0.f, 0.f, 0.f};
  for (; i < n4; i += stride) ((float4*)p)[i] = zz;
}

// ---------------------------------------------------------------------------
// GEMM: C[M,128] = A(bf16 [M,128]) * W^T + bias, W bf16 [128,128] row-major
// ([n][k], i.e. exactly the B=W^T fragment source), fp32 accumulate via
// v_wmma_f32_16x16x32_bf16.
//
// Block = 256 threads (8 waves). W cooperatively staged into LDS (32KB,
// 272B row stride: each bank hit exactly 2x per wave-wide b128 op).
// Each wave owns a 16-row tile and all 128 output columns. The (k0,t) loops
// are flattened into a 32-step pipeline with one-deep register double
// buffering of B fragments and early A prefetch, so ds_loads for step s+1
// are in flight while the WMMA of step s executes.
//
// Fragment layouts per CDNA5 ISA 7.12.2 (wave32):
//   A 16x32 bf16 : lane L row = m0+(L&15); elems 0..7 = K k0+8*(L>=16)..,
//                  elems 8..15 = K k0+16+8*(L>=16)..   (two b128 loads)
//   B 32x16 bf16 : lane L col = n0+(L&15); same K order as A (two b128 loads)
//   C/D 16x16 f32: VGPR j -> row m0 + j + 8*(L>=16), col n0+(L&15)
// ---------------------------------------------------------------------------
__global__ __launch_bounds__(256) void gemm128_bf16_kernel(
    const bf16_t* __restrict__ A, const bf16_t* __restrict__ W,
    const float* __restrict__ bias, float* __restrict__ C, int M) {
  __shared__ bf16_t wlds[FDIM * WSTRIDE];

  // ---- cooperative stage of W into LDS: 2048 x 16B chunks over 256 threads
#pragma unroll
  for (int it = 0; it < 8; ++it) {
    int chunk = threadIdx.x + it * 256;  // 0..2047
    int row = chunk >> 4;                // 0..127
    int part = chunk & 15;               // 16B unit within row
    const uint4 v = *(const uint4*)(W + (size_t)row * FDIM + part * 8);
    *(uint4*)(&wlds[(size_t)row * WSTRIDE + part * 8]) = v;
  }
  __syncthreads();

  const int wave = threadIdx.x >> 5;
  const int lane = threadIdx.x & 31;
  const int m0 = (blockIdx.x * 8 + wave) * 16;
  const int hi = lane >> 4;
  const int lr = lane & 15;

  if (m0 < M) {  // wave-uniform; M is a multiple of 16
    const bf16_t* arow = A + (size_t)(m0 + lr) * FDIM;

    auto ldA = [&](int k0) -> v16bf {
      return mkv16(*(const v8bf*)(arow + k0 + 8 * hi),
                   *(const v8bf*)(arow + k0 + 16 + 8 * hi));
    };
    auto ldB = [&](int t, int k0) -> v16bf {
      const bf16_t* w = &wlds[(size_t)(t * 16 + lr) * WSTRIDE + k0 + 8 * hi];
      return mkv16(*(const v8bf*)(w), *(const v8bf*)(w + 16));
    };

    // bias folded into accumulator init (acc tile column = t*16+lr, uniform
    // across the 8 row-elements of the lane)
    v8f acc[8];
#pragma unroll
    for (int t = 0; t < 8; ++t) {
      const float bv = bias[t * 16 + lr];
#pragma unroll
      for (int j = 0; j < 8; ++j) acc[t][j] = bv;
    }

    v16bf afc = ldA(0);        // current A fragment (k0)
    v16bf afn = afc;           // next A fragment (k0+32)
    v16bf bcur = ldB(0, 0);    // current B fragment
    v16bf bnxt = bcur;         // next B fragment

#pragma unroll
    for (int s = 0; s < 32; ++s) {
      const int k0 = (s >> 3) * 32;
      const int t = s & 7;
      if (t == 0 && k0 < 96) afn = ldA(k0 + 32);  // 8 WMMAs of lead time
      if (s < 31) {
        const int sn = s + 1;
        bnxt = ldB(sn & 7, (sn >> 3) * 32);       // 1 WMMA of lead time
      }
      acc[t] = __builtin_amdgcn_wmma_f32_16x16x32_bf16(
          false, afc, false, bcur, (short)0, acc[t], false, false);
      if (t == 7) afc = afn;
      bcur = bnxt;
    }

#pragma unroll
    for (int t = 0; t < 8; ++t) {
      const int n = t * 16 + lr;
#pragma unroll
      for (int j = 0; j < 8; ++j) {
        C[(size_t)(m0 + 8 * hi + j) * FDIM + n] = acc[t][j];
      }
    }
  }
}

// ---------------------------------------------------------------------------
// Edge phase: one wave per edge, each lane owns 4 consecutive columns.
//   m      = node_g[dst] + node_g[src] + edge_ga[e]
//   sigma  = sigmoid(m);  ssum[dst]  += sigma
//   ssumh[dst] += sigma * node_d[src]
// ---------------------------------------------------------------------------
__global__ __launch_bounds__(256) void edge_kernel(
    const int* __restrict__ srcI, const int* __restrict__ dstI,
    const float* __restrict__ node_g, const float* __restrict__ node_d,
    const float* __restrict__ edge_ga, float* __restrict__ m_out,
    float* __restrict__ ssum, float* __restrict__ ssumh, int E) {
  int e = blockIdx.x * 8 + (threadIdx.x >> 5);
  if (e >= E) return;
  const int lane = threadIdx.x & 31;
  const int c = lane * 4;
  const int s = srcI[e];
  const int d = dstI[e];

  const float4 gs = *(const float4*)(node_g + (size_t)s * FDIM + c);
  const float4 gd = *(const float4*)(node_g + (size_t)d * FDIM + c);
  const float4 eg = *(const float4*)(edge_ga + (size_t)e * FDIM + c);

  float4 mv;
  mv.x = gs.x + gd.x + eg.x;
  mv.y = gs.y + gd.y + eg.y;
  mv.z = gs.z + gd.z + eg.z;
  mv.w = gs.w + gd.w + eg.w;
  *(float4*)(m_out + (size_t)e * FDIM + c) = mv;

  float4 sg;
  sg.x = 1.0f / (1.0f + __expf(-mv.x));
  sg.y = 1.0f / (1.0f + __expf(-mv.y));
  sg.z = 1.0f / (1.0f + __expf(-mv.z));
  sg.w = 1.0f / (1.0f + __expf(-mv.w));

  float* ps = ssum + (size_t)d * FDIM + c;
  atomicAdd(ps + 0, sg.x);
  atomicAdd(ps + 1, sg.y);
  atomicAdd(ps + 2, sg.z);
  atomicAdd(ps + 3, sg.w);

  const float4 nd = *(const float4*)(node_d + (size_t)s * FDIM + c);
  float* ph = ssumh + (size_t)d * FDIM + c;
  atomicAdd(ph + 0, sg.x * nd.x);
  atomicAdd(ph + 1, sg.y * nd.y);
  atomicAdd(ph + 2, sg.z * nd.z);
  atomicAdd(ph + 3, sg.w * nd.w);
}

// ---------------------------------------------------------------------------
// Per-column BN statistics (sum, sumsq). One thread per column, each block
// covers a row chunk, coalesced row-major reads, 2 atomics per thread.
// ---------------------------------------------------------------------------
__global__ __launch_bounds__(128) void colstats_kernel(
    const float* __restrict__ X, float* __restrict__ stats, int M,
    int rowsPerBlk) {
  const int c = threadIdx.x;
  const int r0 = blockIdx.x * rowsPerBlk;
  int r1 = r0 + rowsPerBlk;
  if (r1 > M) r1 = M;
  float s = 0.0f, s2 = 0.0f;
  for (int r = r0; r < r1; ++r) {
    float v = X[(size_t)r * FDIM + c];
    s += v;
    s2 += v * v;
  }
  atomicAdd(&stats[c], s);
  atomicAdd(&stats[FDIM + c], s2);
}

// Fused: h = ssumh/(ssum+eps) + h (h holds src_update(x)), plus BN stats.
__global__ __launch_bounds__(128) void node_combine_kernel(
    float* __restrict__ h, const float* __restrict__ ssum,
    const float* __restrict__ ssumh, float* __restrict__ stats, int M,
    int rowsPerBlk) {
  const int c = threadIdx.x;
  const int r0 = blockIdx.x * rowsPerBlk;
  int r1 = r0 + rowsPerBlk;
  if (r1 > M) r1 = M;
  float s = 0.0f, s2 = 0.0f;
  for (int r = r0; r < r1; ++r) {
    size_t i = (size_t)r * FDIM + c;
    float v = ssumh[i] / (ssum[i] + GATE_EPS) + h[i];
    h[i] = v;
    s += v;
    s2 += v * v;
  }
  atomicAdd(&stats[c], s);
  atomicAdd(&stats[FDIM + c], s2);
}

// stats: [0..127]=sum, [128..255]=sumsq -> writes [256..383]=mu, [384..511]=rstd
__global__ __launch_bounds__(128) void bn_finalize_kernel(
    float* __restrict__ stats, int M) {
  const int c = threadIdx.x;
  const float inv = 1.0f / (float)M;
  const float mu = stats[c] * inv;
  const float var = stats[FDIM + c] * inv - mu * mu;
  stats[256 + c] = mu;
  stats[384 + c] = rsqrtf(var + BN_EPS);
}

// out = silu((H - mu)*rstd*gamma + beta) + res, vectorized by 4
__global__ __launch_bounds__(256) void apply_bn_silu_res_kernel(
    const float* __restrict__ H, const float* __restrict__ res,
    const float* __restrict__ gamma, const float* __restrict__ beta,
    const float* __restrict__ stats, float* __restrict__ out, size_t n4) {
  size_t i = (size_t)blockIdx.x * blockDim.x + threadIdx.x;
  size_t stride = (size_t)gridDim.x * blockDim.x;
  for (; i < n4; i += stride) {
    const int c = (int)(i & 31) * 4;  // column of first element
    const float4 hv = ((const float4*)H)[i];
    const float4 rv = ((const float4*)res)[i];
    const float4 mu = *(const float4*)(stats + 256 + c);
    const float4 rs = *(const float4*)(stats + 384 + c);
    const float4 g = *(const float4*)(gamma + c);
    const float4 b = *(const float4*)(beta + c);
    float4 o;
    float v;
    v = (hv.x - mu.x) * rs.x * g.x + b.x;
    o.x = v * (1.0f / (1.0f + __expf(-v))) + rv.x;
    v = (hv.y - mu.y) * rs.y * g.y + b.y;
    o.y = v * (1.0f / (1.0f + __expf(-v))) + rv.y;
    v = (hv.z - mu.z) * rs.z * g.z + b.z;
    o.z = v * (1.0f / (1.0f + __expf(-v))) + rv.z;
    v = (hv.w - mu.w) * rs.w * g.w + b.w;
    o.w = v * (1.0f / (1.0f + __expf(-v))) + rv.w;
    ((float4*)out)[i] = o;
  }
}

// ---------------------------------------------------------------------------
// Host orchestration
// ---------------------------------------------------------------------------
static inline unsigned gblocks(size_t n, unsigned bs) {
  size_t b = (n + bs - 1) / bs;
  if (b > (size_t)1073741824u) b = 1073741824u;  // kernels are grid-stride
  return (unsigned)b;
}

extern "C" void kernel_launch(void* const* d_in, const int* in_sizes, int n_in,
                              void* d_out, int out_size, void* d_ws,
                              size_t ws_size, hipStream_t stream) {
  const float* x = (const float*)d_in[0];
  const float* y = (const float*)d_in[1];
  const float* z = (const float*)d_in[2];
  const int* eig = (const int*)d_in[3];   // [2,E]  : src | dst
  const int* eilg = (const int*)d_in[4];  // [2,E2] : src | dst

  const int N = in_sizes[0] / FDIM;
  const int E = in_sizes[1] / FDIM;
  const int E2 = in_sizes[2] / FDIM;

  // params flattened leaf-order: per conv (base pb):
  //  +0 src_gate_w  +1 src_gate_b  +2 edge_gate_w  +3 edge_gate_b
  //  +4 src_update_w +5 src_update_b +6 dst_update_w +7 dst_update_b
  //  +8 bn_nodes_g  +9 bn_nodes_b  +10 bn_edges_g  +11 bn_edges_b
  auto pf = [&](int i) { return (const float*)d_in[i]; };

  // ---- workspace layout (256B-aligned pools) ----
  char* wsb = (char*)d_ws;
  size_t off = 0;
  auto alloc = [&](size_t bytes) -> char* {
    char* p = wsb + off;
    off += (bytes + 255) & ~(size_t)255;
    return p;
  };
  const size_t maxNode = (size_t)((N > E) ? N : E);
  size_t maxRows = (size_t)((E > E2) ? E : E2);
  if ((size_t)N > maxRows) maxRows = (size_t)N;

  float* p_nodeg = (float*)alloc(maxNode * FDIM * 4);
  float* p_noded = (float*)alloc(maxNode * FDIM * 4);
  float* p_ssum  = (float*)alloc(maxNode * FDIM * 4);
  float* p_ssumh = (float*)alloc(maxNode * FDIM * 4);
  float* p_m     = (float*)alloc(maxRows * FDIM * 4);  // pre-BN edge messages
  float* p_ega   = (float*)alloc(maxRows * FDIM * 4);  // edge_gate(edge_attr)
  float* p_mout  = (float*)alloc((size_t)E * FDIM * 4);  // conv1 edge output
  bf16_t* p_bf   = (bf16_t*)alloc(maxRows * FDIM * 2);   // bf16 staging
  bf16_t* p_wb   = (bf16_t*)alloc(8 * FDIM * FDIM * 2);  // bf16 weights
  float* p_stats = (float*)alloc(1024 * 4);  // [node 512 | edge 512]
  (void)ws_size; (void)n_in; (void)out_size;

  float* out_x = (float*)d_out;
  float* out_y = out_x + (size_t)N * FDIM;
  float* out_z = out_y + (size_t)E * FDIM;

  // ---- convert 8 weight matrices to bf16 (order per conv: sg, eg, su, du) ----
  const size_t WSZ = (size_t)FDIM * FDIM;
  const int widx[4] = {0, 2, 4, 6};
  for (int cv = 0; cv < 2; ++cv) {
    int pb = 5 + cv * 12;
    for (int j = 0; j < 4; ++j) {
      f32_to_bf16_kernel<<<gblocks(WSZ / 4, 256), 256, 0, stream>>>(
          pf(pb + widx[j]), p_wb + (cv * 4 + j) * WSZ, WSZ / 4);
    }
  }

  auto run_conv = [&](int rowsX, const float* Xin, int rowsE, const float* Ein,
                      const int* srcI, const int* dstI, int pb,
                      const bf16_t* Wsg, const bf16_t* Weg, const bf16_t* Wsu,
                      const bf16_t* Wdu, float* hbuf, float* out_node,
                      float* out_edge) {
    const size_t nX = (size_t)rowsX * FDIM;
    const size_t nE = (size_t)rowsE * FDIM;

    // node-side GEMMs (src_gate, dst_update, src_update->hbuf)
    f32_to_bf16_kernel<<<gblocks(nX / 4, 256), 256, 0, stream>>>(Xin, p_bf,
                                                                 nX / 4);
    const unsigned gx = (unsigned)((rowsX + 127) / 128);
    gemm128_bf16_kernel<<<gx, 256, 0, stream>>>(p_bf, Wsg, pf(pb + 1), p_nodeg, rowsX);
    gemm128_bf16_kernel<<<gx, 256, 0, stream>>>(p_bf, Wdu, pf(pb + 7), p_noded, rowsX);
    gemm128_bf16_kernel<<<gx, 256, 0, stream>>>(p_bf, Wsu, pf(pb + 5), hbuf, rowsX);

    // edge_gate GEMM
    f32_to_bf16_kernel<<<gblocks(nE / 4, 256), 256, 0, stream>>>(Ein, p_bf,
                                                                 nE / 4);
    const unsigned ge = (unsigned)((rowsE + 127) / 128);
    gemm128_bf16_kernel<<<ge, 256, 0, stream>>>(p_bf, Weg, pf(pb + 3), p_ega, rowsE);

    // segment sums
    zero_f32_kernel<<<gblocks(nX / 4, 256), 256, 0, stream>>>(p_ssum, nX / 4);
    zero_f32_kernel<<<gblocks(nX / 4, 256), 256, 0, stream>>>(p_ssumh, nX / 4);
    edge_kernel<<<(unsigned)((rowsE + 7) / 8), 256, 0, stream>>>(
        srcI, dstI, p_nodeg, p_noded, p_ega, p_m, p_ssum, p_ssumh, rowsE);

    // batchnorm statistics (node over hbuf, edge over p_m)
    zero_f32_kernel<<<1, 256, 0, stream>>>(p_stats, 256);
    const int RPB = 1024;
    node_combine_kernel<<<(unsigned)((rowsX + RPB - 1) / RPB), 128, 0, stream>>>(
        hbuf, p_ssum, p_ssumh, p_stats, rowsX, RPB);
    colstats_kernel<<<(unsigned)((rowsE + RPB - 1) / RPB), 128, 0, stream>>>(
        p_m, p_stats + 512, rowsE, RPB);
    bn_finalize_kernel<<<1, 128, 0, stream>>>(p_stats, rowsX);
    bn_finalize_kernel<<<1, 128, 0, stream>>>(p_stats + 512, rowsE);

    // out = silu(bn(.)) + residual
    apply_bn_silu_res_kernel<<<gblocks(nX / 4, 256), 256, 0, stream>>>(
        hbuf, Xin, pf(pb + 8), pf(pb + 9), p_stats, out_node, nX / 4);
    apply_bn_silu_res_kernel<<<gblocks(nE / 4, 256), 256, 0, stream>>>(
        p_m, Ein, pf(pb + 10), pf(pb + 11), p_stats + 512, out_edge, nE / 4);
  };

  // conv1: nodes = x (N), edges = y (E); node out -> x_out, edge out -> m_out
  run_conv(N, x, E, y, eig, eig + E, 5,
           p_wb + 0 * WSZ, p_wb + 1 * WSZ, p_wb + 2 * WSZ, p_wb + 3 * WSZ,
           out_x, out_x, p_mout);

  // conv2: nodes = m_out (E), edges = z (E2); node out -> y_out, edge -> z_out
  run_conv(E, p_mout, E2, z, eilg, eilg + E2, 17,
           p_wb + 4 * WSZ, p_wb + 5 * WSZ, p_wb + 6 * WSZ, p_wb + 7 * WSZ,
           out_y, out_y, out_z);
}
